// MultiHeadAttention_8770323219084
// MI455X (gfx1250) — matrix operand order repaired and checked by reading, not compiled, not run on hardware
//
#include <hip/hip_runtime.h>
#include <hip/hip_bf16.h>

typedef __attribute__((ext_vector_type(16))) _Float16 v16h;
typedef __attribute__((ext_vector_type(8)))  float    v8f;
typedef __attribute__((ext_vector_type(4)))  int      v4i;

#define BATCH  4
#define SEQ    2048
#define DMODEL 1024
#define NHEAD  16
#define DHEAD  64

#if __has_builtin(__builtin_amdgcn_global_load_async_to_lds_b128) && \
    __has_builtin(__builtin_amdgcn_s_wait_asynccnt)
#define USE_ASYNC_LDS 1
typedef __attribute__((address_space(1))) v4i* gas1_v4i;
typedef __attribute__((address_space(3))) v4i* gas3_v4i;
#else
#define USE_ASYNC_LDS 0
#endif

// ---------------- WMMA fragment loaders (per ISA 16-bit layouts) ----------------
// A-matrix 16x32: lane<16 holds K=0..7 (h0..7) and K=16..23 (h8..15); lane>=16: K=8..15, K=24..31.
// Caller passes p = &tile[row*stride + (lane>>4)*8].
__device__ __forceinline__ v16h frag_a_from_lds(const _Float16* p) {
  v16h a;
#pragma unroll
  for (int i = 0; i < 8; ++i) a[i] = p[i];
#pragma unroll
  for (int i = 0; i < 8; ++i) a[8 + i] = p[16 + i];
  return a;
}
// B-matrix 32x16: lane&15 = column; lane<16 holds K=0..7, K=16..23; lane>=16 K=8..15, K=24..31.
// Caller passes p = &tileT[col*stride + (lane>>4)*8].
__device__ __forceinline__ v16h frag_b_from_lds(const _Float16* p) {
  v16h b;
#pragma unroll
  for (int i = 0; i < 8; ++i) b[i] = p[i];
#pragma unroll
  for (int i = 0; i < 8; ++i) b[8 + i] = p[16 + i];
  return b;
}

// ---------------- Kernel 1: fused QKV projection (fp32 -> f16 once) ----------------
// out[row, col] = sum_d x[row,d] * W[col,d]   (x @ W^T), stored f16 head-major [B,H,S,DH]
// Double-buffered LDS + register-staged global pipeline.
__global__ __launch_bounds__(256)
void qkv_proj_kernel(const float* __restrict__ x,
                     const float* __restrict__ Wq,
                     const float* __restrict__ Wk,
                     const float* __restrict__ Wv,
                     _Float16* __restrict__ Qh,
                     _Float16* __restrict__ Kh,
                     _Float16* __restrict__ Vh)
{
  const float* W   = (blockIdx.z == 0) ? Wq : (blockIdx.z == 1) ? Wk : Wv;
  _Float16*   Out  = (blockIdx.z == 0) ? Qh : (blockIdx.z == 1) ? Kh : Vh;

  constexpr int AST = 40;                       // padded LDS stride (halves)
  __shared__ _Float16 Alds[2][128 * AST];
  __shared__ _Float16 Blds[2][128 * AST];

  const int t     = threadIdx.x;
  const int lane  = t & 31;
  const int wave  = t >> 5;
  const int waveM = wave >> 2;                  // 0..1  -> 64 rows each
  const int waveN = wave & 3;                   // 0..3  -> 32 cols each
  const int mBase = blockIdx.y * 128;
  const int nBase = blockIdx.x * 128;

  v8f zero = {};
  v8f acc[4][2];
#pragma unroll
  for (int i = 0; i < 4; ++i)
#pragma unroll
    for (int j = 0; j < 2; ++j) acc[i][j] = zero;

  const int rA    = lane & 15;
  const int kOffA = (lane >> 4) * 8;
  const int kOffB = (lane >> 4) * 8;

  float4 fa[4], fb[4];
  auto load_tiles = [&](int k0) {
#pragma unroll
    for (int i = 0; i < 4; ++i) {
      int idx = t + i * 256;
      int row = idx >> 3;
      int c4  = (idx & 7) << 2;
      fa[i] = *(const float4*)(x + (size_t)(mBase + row) * DMODEL + k0 + c4);
      fb[i] = *(const float4*)(W + (size_t)(nBase + row) * DMODEL + k0 + c4);
    }
  };
  auto stash_tiles = [&](int buf) {
#pragma unroll
    for (int i = 0; i < 4; ++i) {
      int idx = t + i * 256;
      int row = idx >> 3;
      int c4  = (idx & 7) << 2;
      _Float16* da = &Alds[buf][row * AST + c4];
      da[0] = (_Float16)fa[i].x; da[1] = (_Float16)fa[i].y;
      da[2] = (_Float16)fa[i].z; da[3] = (_Float16)fa[i].w;
      _Float16* db = &Blds[buf][row * AST + c4];
      db[0] = (_Float16)fb[i].x; db[1] = (_Float16)fb[i].y;
      db[2] = (_Float16)fb[i].z; db[3] = (_Float16)fb[i].w;
    }
  };

  load_tiles(0);
  stash_tiles(0);

  constexpr int NK = DMODEL / 32;               // 32 K-steps
  for (int ks = 0; ks < NK; ++ks) {
    __syncthreads();
    const int  cur  = ks & 1;
    const bool more = (ks + 1) < NK;
    if (more) load_tiles((ks + 1) * 32);        // prefetch next tile into regs

    v16h af[4], bf[2];
#pragma unroll
    for (int mt = 0; mt < 4; ++mt)
      af[mt] = frag_a_from_lds(&Alds[cur][(waveM * 64 + mt * 16 + rA) * AST + kOffA]);
#pragma unroll
    for (int nt = 0; nt < 2; ++nt)
      bf[nt] = frag_b_from_lds(&Blds[cur][(waveN * 32 + nt * 16 + rA) * AST + kOffB]);
#pragma unroll
    for (int mt = 0; mt < 4; ++mt)
#pragma unroll
      for (int nt = 0; nt < 2; ++nt)
        acc[mt][nt] = __builtin_amdgcn_wmma_f32_16x16x32_f16(
            false, af[mt], false, bf[nt], (short)0, acc[mt][nt], false, false);

    if (more) stash_tiles(cur ^ 1);             // convert+store next tile (other buffer)
  }

  // C layout: VGPR r -> row r + (lane>>4)*8 in tile, col = lane&15
  const int r0 = (lane >> 4) * 8;
  const int cn = lane & 15;
#pragma unroll
  for (int mt = 0; mt < 4; ++mt)
#pragma unroll
    for (int nt = 0; nt < 2; ++nt)
#pragma unroll
      for (int r = 0; r < 8; ++r) {
        int row = mBase + waveM * 64 + mt * 16 + r + r0;   // token row (b*S+s)
        int col = nBase + waveN * 32 + nt * 16 + cn;       // d_model index
        int b  = row >> 11;
        int s  = row & (SEQ - 1);
        int h  = col >> 6;
        int dh = col & (DHEAD - 1);
        Out[(((size_t)b * NHEAD + h) * SEQ + s) * DHEAD + dh] = (_Float16)acc[mt][nt][r];
      }
}

// ---------------- Kernel 2: causal flash attention, f16 WMMA, fp32 softmax ----------------
__global__ __launch_bounds__(256)
void attn_kernel(const _Float16* __restrict__ Qh,
                 const _Float16* __restrict__ Kh,
                 const _Float16* __restrict__ Vh,
                 _Float16* __restrict__ AO)
{
  constexpr int QST = 72, KST = 72, PST = 72;   // padded strides (halves)
  __shared__ _Float16 Qlds[128 * QST];          // [query][dh]
  __shared__ _Float16 Klds[64 * KST];           // [key][dh]
  __shared__ _Float16 VTlds[64 * KST];          // [dh][key] (transposed)
  __shared__ _Float16 Plds[128 * PST];          // [query][key] probabilities

  const int t     = threadIdx.x;
  const int lane  = t & 31;
  const int wave  = t >> 5;                     // 8 waves x 16 query rows = 128
  const int qTile = blockIdx.x;
  const int bh    = blockIdx.y;                 // b*NHEAD + h
  const int qBase = qTile * 128;

  const _Float16* Qg = Qh + (size_t)bh * SEQ * DHEAD;
  const _Float16* Kg = Kh + (size_t)bh * SEQ * DHEAD;
  const _Float16* Vg = Vh + (size_t)bh * SEQ * DHEAD;

  // stage Q tile once: 128x64 halves = 1024 16B chunks, 4/thread
#pragma unroll
  for (int i = 0; i < 4; ++i) {
    int idx = t + i * 256;
    int row = idx >> 3;
    int c8  = (idx & 7) << 3;
#if USE_ASYNC_LDS
    __builtin_amdgcn_global_load_async_to_lds_b128(
        (gas1_v4i)(void*)&Qg[(size_t)(qBase + row) * DHEAD + c8],
        (gas3_v4i)(void*)&Qlds[row * QST + c8], 0, 0);
#else
    *(uint4*)&Qlds[row * QST + c8] =
        *(const uint4*)&Qg[(size_t)(qBase + row) * DHEAD + c8];
#endif
  }

  v8f zero = {};
  float m_i[8], l_i[8];
  v8f o[4];
#pragma unroll
  for (int r = 0; r < 8; ++r) { m_i[r] = -3.0e38f; l_i[r] = 0.f; }
#pragma unroll
  for (int d = 0; d < 4; ++d) o[d] = zero;

  const float scale = 0.125f;                   // 1/sqrt(64)
  const int rA    = lane & 15;
  const int r0    = (lane >> 4) * 8;
  const int kOffA = (lane >> 4) * 8;
  const int kOffB = (lane >> 4) * 8;

  const int nkb = 2 * qTile + 2;                // 64-wide key blocks up to the diagonal
  for (int kb = 0; kb < nkb; ++kb) {
    const int kBase = kb * 64;
    __syncthreads();                            // protect K/V LDS reuse
    // K tile 64x64: 512 chunks, 2/thread (async DMA to LDS when available)
#pragma unroll
    for (int i = 0; i < 2; ++i) {
      int idx = t + i * 256;
      int row = idx >> 3;
      int c8  = (idx & 7) << 3;
#if USE_ASYNC_LDS
      __builtin_amdgcn_global_load_async_to_lds_b128(
          (gas1_v4i)(void*)&Kg[(size_t)(kBase + row) * DHEAD + c8],
          (gas3_v4i)(void*)&Klds[row * KST + c8], 0, 0);
#else
      *(uint4*)&Klds[row * KST + c8] =
          *(const uint4*)&Kg[(size_t)(kBase + row) * DHEAD + c8];
#endif
    }
    // V tile transposed: 4096 halves, 16/thread (scatter -> plain DS stores)
#pragma unroll
    for (int i = 0; i < 16; ++i) {
      int idx = t + i * 256;
      int s   = idx >> 6;
      int dh  = idx & 63;
      VTlds[dh * KST + s] = Vg[(size_t)(kBase + s) * DHEAD + dh];
    }
#if USE_ASYNC_LDS
    __builtin_amdgcn_s_wait_asynccnt(0);
#endif
    __syncthreads();

    // scores: 16 queries x 64 keys per wave, K-dim = DH = 64 (2 wmma steps)
    v8f sc[4];
#pragma unroll
    for (int nt = 0; nt < 4; ++nt) sc[nt] = zero;
#pragma unroll
    for (int ks = 0; ks < 2; ++ks) {
      v16h aq = frag_a_from_lds(&Qlds[(wave * 16 + rA) * QST + ks * 32 + kOffA]);
#pragma unroll
      for (int nt = 0; nt < 4; ++nt) {
        v16h bk = frag_b_from_lds(&Klds[(nt * 16 + rA) * KST + ks * 32 + kOffB]);
        sc[nt] = __builtin_amdgcn_wmma_f32_16x16x32_f16(
            false, aq, false, bk, (short)0, sc[nt], false, false);
      }
    }

    // scale + causal mask (only last two blocks can violate causality)
    const bool needMask = (kb >= 2 * qTile);
#pragma unroll
    for (int nt = 0; nt < 4; ++nt) {
      int key = kBase + nt * 16 + rA;
#pragma unroll
      for (int r = 0; r < 8; ++r) {
        float sv = sc[nt][r] * scale;
        if (needMask) {
          int q = qBase + wave * 16 + r + r0;
          if (key > q) sv = -3.0e38f;
        }
        sc[nt][r] = sv;
      }
    }

    // online softmax: row lives across one 16-lane half-group -> xor-shuffle reduce
    float mnew[8], alpha[8], rs[8];
#pragma unroll
    for (int r = 0; r < 8; ++r) {
      float v = sc[0][r];
#pragma unroll
      for (int nt = 1; nt < 4; ++nt) v = fmaxf(v, sc[nt][r]);
#pragma unroll
      for (int off = 8; off >= 1; off >>= 1) v = fmaxf(v, __shfl_xor(v, off, 32));
      mnew[r]  = fmaxf(m_i[r], v);
      alpha[r] = __expf(m_i[r] - mnew[r]);
      rs[r]    = 0.f;
    }
#pragma unroll
    for (int nt = 0; nt < 4; ++nt) {
#pragma unroll
      for (int r = 0; r < 8; ++r) {
        float p = __expf(sc[nt][r] - mnew[r]);
        rs[r] += p;
        Plds[(wave * 16 + r + r0) * PST + nt * 16 + rA] = (_Float16)p;
      }
    }
#pragma unroll
    for (int r = 0; r < 8; ++r) {
#pragma unroll
      for (int off = 8; off >= 1; off >>= 1) rs[r] += __shfl_xor(rs[r], off, 32);
      l_i[r] = l_i[r] * alpha[r] + rs[r];
      m_i[r] = mnew[r];
    }
#pragma unroll
    for (int d = 0; d < 4; ++d)
#pragma unroll
      for (int r = 0; r < 8; ++r) o[d][r] *= alpha[r];

    // O += P @ V  (K-dim = 64 keys, 2 wmma steps; P region is wave-private in LDS)
#pragma unroll
    for (int ks = 0; ks < 2; ++ks) {
      v16h ap = frag_a_from_lds(&Plds[(wave * 16 + rA) * PST + ks * 32 + kOffA]);
#pragma unroll
      for (int d = 0; d < 4; ++d) {
        v16h bv = frag_b_from_lds(&VTlds[(d * 16 + rA) * KST + ks * 32 + kOffB]);
        o[d] = __builtin_amdgcn_wmma_f32_16x16x32_f16(
            false, ap, false, bv, (short)0, o[d], false, false);
      }
    }
  }

  // normalize and write attn output as f16 [B,S,D] (D index = h*64+dh)
  const int cn = lane & 15;
  const int b  = bh >> 4;
  const int h  = bh & 15;
#pragma unroll
  for (int d = 0; d < 4; ++d)
#pragma unroll
    for (int r = 0; r < 8; ++r) {
      int q  = qBase + wave * 16 + r + r0;
      int dh = d * 16 + cn;
      float outv = o[d][r] / l_i[r];
      AO[((size_t)b * SEQ + q) * DMODEL + h * DHEAD + dh] = (_Float16)outv;
    }
}

// ---------------- Kernel 3: output projection (f16 A, fp32 W converted, fp32 out) ----------------
__global__ __launch_bounds__(256)
void out_proj_kernel(const _Float16* __restrict__ A,
                     const float* __restrict__ Wo,
                     float* __restrict__ Out)
{
  constexpr int AST = 40;
  __shared__ _Float16 Alds[2][128 * AST];
  __shared__ _Float16 Blds[2][128 * AST];

  const int t     = threadIdx.x;
  const int lane  = t & 31;
  const int wave  = t >> 5;
  const int waveM = wave >> 2;
  const int waveN = wave & 3;
  const int mBase = blockIdx.y * 128;
  const int nBase = blockIdx.x * 128;

  v8f zero = {};
  v8f acc[4][2];
#pragma unroll
  for (int i = 0; i < 4; ++i)
#pragma unroll
    for (int j = 0; j < 2; ++j) acc[i][j] = zero;

  const int rA    = lane & 15;
  const int kOffA = (lane >> 4) * 8;
  const int kOffB = (lane >> 4) * 8;

  uint4  ua[2];
  float4 fb[4];
  auto load_tiles = [&](int k0) {
#pragma unroll
    for (int i = 0; i < 2; ++i) {              // A: 512 16B chunks, 2/thread
      int idx = t + i * 256;
      int row = idx >> 2;
      int c8  = (idx & 3) << 3;
      ua[i] = *(const uint4*)&A[(size_t)(mBase + row) * DMODEL + k0 + c8];
    }
#pragma unroll
    for (int i = 0; i < 4; ++i) {              // B: 1024 float4 chunks, 4/thread
      int idx = t + i * 256;
      int row = idx >> 3;
      int c4  = (idx & 7) << 2;
      fb[i] = *(const float4*)(Wo + (size_t)(nBase + row) * DMODEL + k0 + c4);
    }
  };
  auto stash_tiles = [&](int buf) {
#pragma unroll
    for (int i = 0; i < 2; ++i) {
      int idx = t + i * 256;
      int row = idx >> 2;
      int c8  = (idx & 3) << 3;
      *(uint4*)&Alds[buf][row * AST + c8] = ua[i];
    }
#pragma unroll
    for (int i = 0; i < 4; ++i) {
      int idx = t + i * 256;
      int row = idx >> 3;
      int c4  = (idx & 7) << 2;
      _Float16* db = &Blds[buf][row * AST + c4];
      db[0] = (_Float16)fb[i].x; db[1] = (_Float16)fb[i].y;
      db[2] = (_Float16)fb[i].z; db[3] = (_Float16)fb[i].w;
    }
  };

  load_tiles(0);
  stash_tiles(0);

  constexpr int NK = DMODEL / 32;
  for (int ks = 0; ks < NK; ++ks) {
    __syncthreads();
    const int  cur  = ks & 1;
    const bool more = (ks + 1) < NK;
    if (more) load_tiles((ks + 1) * 32);

    v16h af[4], bf2[2];
#pragma unroll
    for (int mt = 0; mt < 4; ++mt)
      af[mt] = frag_a_from_lds(&Alds[cur][(waveM * 64 + mt * 16 + rA) * AST + kOffA]);
#pragma unroll
    for (int nt = 0; nt < 2; ++nt)
      bf2[nt] = frag_b_from_lds(&Blds[cur][(waveN * 32 + nt * 16 + rA) * AST + kOffB]);
#pragma unroll
    for (int mt = 0; mt < 4; ++mt)
#pragma unroll
      for (int nt = 0; nt < 2; ++nt)
        acc[mt][nt] = __builtin_amdgcn_wmma_f32_16x16x32_f16(
            false, af[mt], false, bf2[nt], (short)0, acc[mt][nt], false, false);

    if (more) stash_tiles(cur ^ 1);
  }

  const int r0 = (lane >> 4) * 8;
  const int cn = lane & 15;
#pragma unroll
  for (int mt = 0; mt < 4; ++mt)
#pragma unroll
    for (int nt = 0; nt < 2; ++nt)
#pragma unroll
      for (int r = 0; r < 8; ++r) {
        int row = mBase + waveM * 64 + mt * 16 + r + r0;
        int col = nBase + waveN * 32 + nt * 16 + cn;
        Out[(size_t)row * DMODEL + col] = acc[mt][nt][r];
      }
}

// ---------------- host-side launch ----------------
extern "C" void kernel_launch(void* const* d_in, const int* in_sizes, int n_in,
                              void* d_out, int out_size, void* d_ws, size_t ws_size,
                              hipStream_t stream) {
  const float* x  = (const float*)d_in[0];
  const float* Wq = (const float*)d_in[1];
  const float* Wk = (const float*)d_in[2];
  const float* Wv = (const float*)d_in[3];
  const float* Wo = (const float*)d_in[4];
  float* out = (float*)d_out;

  const size_t elems = (size_t)BATCH * SEQ * DMODEL;   // 8Mi
  _Float16* Qh = (_Float16*)d_ws;                      // 16 MB each
  _Float16* Kh = Qh + elems;
  _Float16* Vh = Kh + elems;
  _Float16* AO = Vh + elems;                           // total 64 MB of d_ws

  dim3 gProj(DMODEL / 128, (BATCH * SEQ) / 128, 3);
  qkv_proj_kernel<<<gProj, 256, 0, stream>>>(x, Wq, Wk, Wv, Qh, Kh, Vh);

  dim3 gAttn(SEQ / 128, BATCH * NHEAD, 1);
  attn_kernel<<<gAttn, 256, 0, stream>>>(Qh, Kh, Vh, AO);

  dim3 gOut(DMODEL / 128, (BATCH * SEQ) / 128, 1);
  out_proj_kernel<<<gOut, 256, 0, stream>>>(AO, Wo, out);
}
